// ConvMultiHeadAttn_1786706395826
// MI455X (gfx1250) — matrix operand order repaired and checked by reading, not compiled, not run on hardware
//
#include <hip/hip_runtime.h>
#include <hip/hip_bf16.h>

// ---------------------------------------------------------------------------
// Problem constants: B=16, L=512, D=1024, H=16, DH=64
// ---------------------------------------------------------------------------
#define BB   16
#define LL   512
#define DD   1024
#define HH   16
#define DHH  64
#define MTOT (BB * LL)      // 8192 rows
#define NEGV (-1e9f)

typedef __attribute__((ext_vector_type(16))) __bf16 v16bf;
typedef __attribute__((ext_vector_type(8)))  float  v8f;

union AFrag {
    v16bf v;
    uint4 q[2];
    unsigned short u[16];
};

static __device__ __forceinline__ unsigned short f2bf(float f) {
    // round-to-nearest-even f32 -> bf16
    unsigned int u = __float_as_uint(f);
    u += 0x7FFFu + ((u >> 16) & 1u);
    return (unsigned short)(u >> 16);
}

static __device__ __forceinline__ v8f wmma_bf16(const AFrag& a, const AFrag& b, v8f c) {
    // D = A(16x32 bf16) * B(32x16 bf16) + C(16x16 f32)
    return __builtin_amdgcn_wmma_f32_16x16x32_bf16(
        /*neg_a=*/false, a.v, /*neg_b=*/false, b.v,
        /*c_mod=*/(short)0, c, /*reuse_a=*/false, /*reuse_b=*/false);
}

// ---------------------------------------------------------------------------
// Kernel 1: f32 -> bf16 conversion (vectorized, grid-stride), row-major copy.
// ---------------------------------------------------------------------------
__global__ __launch_bounds__(256) void cvt_bf16(const float* __restrict__ src,
                                                unsigned short* __restrict__ dst,
                                                int n4) {
    int idx = blockIdx.x * blockDim.x + threadIdx.x;
    int stride = gridDim.x * blockDim.x;
    for (int i = idx; i < n4; i += stride) {
        float4 f = ((const float4*)src)[i];
        uint2 o;
        o.x = (unsigned int)f2bf(f.x) | ((unsigned int)f2bf(f.y) << 16);
        o.y = (unsigned int)f2bf(f.z) | ((unsigned int)f2bf(f.w) << 16);
        ((uint2*)dst)[i] = o;
    }
}

// ---------------------------------------------------------------------------
// Kernel 2: pack weight matrix W[K=1024][N] (f32 row-major) into B-fragment
// order: dst[((kt*(N/16)+ntt)*32 + lane)*16 + j]  with
//   K index = kt*32 + ((j&8)<<1) + ((lane&16)>>1) + (j&7),  N index = ntt*16 + (lane&15)
// ---------------------------------------------------------------------------
__global__ __launch_bounds__(256) void pack_b(const float* __restrict__ W,
                                              unsigned short* __restrict__ dst,
                                              int N) {
    int tid  = blockIdx.x * blockDim.x + threadIdx.x;
    int lane = tid & 31;
    int tile = tid >> 5;
    int nt16 = N >> 4;
    int kt   = tile / nt16;
    int ntt  = tile - kt * nt16;
    int off  = lane & 16;             // 0 or 16
    int n    = ntt * 16 + (lane & 15);
    AFrag f;
#pragma unroll
    for (int j = 0; j < 16; j++)
        f.u[j] = f2bf(W[(size_t)(kt * 32 + ((j & 8) << 1) + (off >> 1) + (j & 7)) * N + n]);
    uint4* d = (uint4*)(dst + (size_t)tid * 16);
    d[0] = f.q[0];
    d[1] = f.q[1];
}

// ---------------------------------------------------------------------------
// GEMM helpers: 32x64 wave tile, register double-buffered mainloop.
// ---------------------------------------------------------------------------
static __device__ __forceinline__ void load_a2(AFrag a[2],
                                               const unsigned short* abase0,
                                               const unsigned short* abase1,
                                               int k0, int hi8) {
    a[0].q[0] = *(const uint4*)(abase0 + k0 + hi8);
    a[0].q[1] = *(const uint4*)(abase0 + k0 + 16 + hi8);
    a[1].q[0] = *(const uint4*)(abase1 + k0 + hi8);
    a[1].q[1] = *(const uint4*)(abase1 + k0 + 16 + hi8);
}

static __device__ __forceinline__ void load_b4(AFrag b[4],
                                               const unsigned short* wp,
                                               int kt, int nt16, int nt64,
                                               int lane) {
    const unsigned short* bb =
        wp + ((size_t)((kt * nt16 + nt64 * 4) * 32 + lane)) * 16;
#pragma unroll
    for (int t = 0; t < 4; t++) {
        b[t].q[0] = ((const uint4*)bb)[0];
        b[t].q[1] = ((const uint4*)bb)[1];
        bb += 32 * 16;
    }
}

static __device__ __forceinline__ void mma_8(v8f acc[2][4],
                                             const AFrag a[2],
                                             const AFrag b[4]) {
#pragma unroll
    for (int mh = 0; mh < 2; mh++)
#pragma unroll
        for (int t = 0; t < 4; t++)
            acc[mh][t] = wmma_bf16(a[mh], b[t], acc[mh][t]);
}

// ---------------------------------------------------------------------------
// Kernel 3: QKV GEMM.  C[8192,4096] = Xbf[8192,1024] @ Wqkv_pack.
// One wave computes a 32x64 tile (8 WMMAs / k-step), double-buffered.
// Epilogue scatters bf16 into attention-ready layouts:
//   q    : [b][h][l][dh] row-major         (A-matrix for scores)
//   k1/k2: B-frag pack of K^T (dh x l)     (B-matrix for scores)
//   v    : B-frag pack of V   (l  x dh)    (B-matrix for P@V)
// ---------------------------------------------------------------------------
__global__ __launch_bounds__(128) void gemm_qkv(const unsigned short* __restrict__ xb,
                                                const unsigned short* __restrict__ wp,
                                                unsigned short* __restrict__ qbf,
                                                unsigned short* __restrict__ k1p,
                                                unsigned short* __restrict__ k2p,
                                                unsigned short* __restrict__ vp) {
    int lane = threadIdx.x & 31;
    int wv   = threadIdx.x >> 5;
    int task = blockIdx.x * 4 + wv;   // 16384 tasks = 256 mt * 64 nt64
    int mt   = task >> 6;
    int nt64 = task & 63;
    int hi   = (lane >= 16) ? 1 : 0;
    int hi8  = hi * 8;
    const unsigned short* abase0 = xb + (size_t)(mt * 32 + (lane & 15)) * DD;
    const unsigned short* abase1 = abase0 + (size_t)16 * DD;

    v8f acc[2][4] = {};
    AFrag a0[2], b0[4], a1[2], b1[4];
    load_a2(a0, abase0, abase1, 0, hi8);
    load_b4(b0, wp, 0, 256, nt64, lane);
    for (int k0 = 0; k0 < DD; k0 += 64) {
        load_a2(a1, abase0, abase1, k0 + 32, hi8);
        load_b4(b1, wp, (k0 >> 5) + 1, 256, nt64, lane);
        mma_8(acc, a0, b0);
        if (k0 + 64 < DD) {
            load_a2(a0, abase0, abase1, k0 + 64, hi8);
            load_b4(b0, wp, (k0 >> 5) + 2, 256, nt64, lane);
        }
        mma_8(acc, a1, b1);
    }

    int region = nt64 >> 4;           // 0=q 1=k1 2=k2 3=v
#pragma unroll
    for (int mh = 0; mh < 2; mh++) {
#pragma unroll
        for (int t = 0; t < 4; t++) {
            int cg = nt64 * 64 + t * 16 + (lane & 15);  // global col in [0,4096)
            int c  = cg & (DD - 1);                     // col within region
            int h  = c >> 6, dh = c & 63;
#pragma unroll
            for (int i = 0; i < 8; i++) {
                int mg = mt * 32 + mh * 16 + i + hi8;
                int b  = mg >> 9, l = mg & (LL - 1);
                int bh = b * HH + h;
                unsigned short val = f2bf(acc[mh][t][i]);
                if (region == 0) {
                    qbf[((size_t)bh * LL + l) * DHH + dh] = val;
                } else if (region == 3) {
                    int kc = l >> 5, kd = l & 31;
                    int lp = ((kd & 8) << 1) + (dh & 15);
                    int j  = (kd & 7) + ((kd & 16) >> 1), nt2 = dh >> 4;
                    vp[((((size_t)bh * 16 + kc) * 4 + nt2) * 32 + lp) * 16 + j] = val;
                } else {
                    int kt = dh >> 5, kd = dh & 31;
                    int lp = ((kd & 8) << 1) + (l & 15);
                    int j  = (kd & 7) + ((kd & 16) >> 1), nt = l >> 4;
                    size_t idx = ((((size_t)bh * 2 + kt) * 32 + nt) * 32 + lp) * 16 + j;
                    if (region == 1) k1p[idx] = val; else k2p[idx] = val;
                }
            }
        }
    }
}

// ---------------------------------------------------------------------------
// Kernel 4: attention. One wave per (b, h, 16-row q tile). Flash-style online
// softmax over 32-key chunks; dual score WMMAs (self/others) with per-element
// qmask select, key-mask fill, Gaussian bias; P relayout C->A via LDS.
// V-fragment loads are hoisted above the softmax math to overlap latency.
// ---------------------------------------------------------------------------
__global__ __launch_bounds__(128) void attn_kernel(
        const unsigned short* __restrict__ qbf,
        const unsigned short* __restrict__ k1p,
        const unsigned short* __restrict__ k2p,
        const unsigned short* __restrict__ vp,
        const long long* __restrict__ mask,
        const long long* __restrict__ qmask,
        const float* __restrict__ shiftp,
        const float* __restrict__ biasp,
        const int* __restrict__ useg,
        unsigned short* __restrict__ aout) {
    __shared__ __align__(16) unsigned short lds[4][16][32];  // per-wave 16x32 P

    int lane = threadIdx.x & 31;
    int wv   = threadIdx.x >> 5;
    int task = blockIdx.x * 4 + wv;   // 8192 = 16b * 16h * 32 qtiles
    int qt   = task & 31;
    int h    = (task >> 5) & 15;
    int b    = task >> 9;
    int hi   = (lane >= 16) ? 1 : 0;
    int hi8  = hi * 8;
    int bh   = b * HH + h;

    float shift = shiftp[0];
    float bias  = biasp[0];
    int   g     = useg[0];

    // Q A-fragments (K = DH = 64 -> two 32-wide chunks)
    AFrag qa[2];
    {
        const unsigned short* qbase =
            qbf + ((size_t)bh * LL + qt * 16 + (lane & 15)) * DHH;
#pragma unroll
        for (int kt = 0; kt < 2; kt++) {
            qa[kt].q[0] = *(const uint4*)(qbase + kt * 32 + hi8);
            qa[kt].q[1] = *(const uint4*)(qbase + kt * 32 + 16 + hi8);
        }
    }

    // qmask for this lane's 8 C-fragment rows
    int qmq[8];
#pragma unroll
    for (int i = 0; i < 8; i++)
        qmq[i] = (int)qmask[(size_t)b * LL + qt * 16 + hi8 + i];

    float runmax[8], runsum[8];
#pragma unroll
    for (int i = 0; i < 8; i++) { runmax[i] = -3.0e38f; runsum[i] = 0.f; }
    v8f O[4] = {};

    for (int kc = 0; kc < 16; kc++) {          // 32 keys per chunk
        float S[2][8];
#pragma unroll
        for (int t = 0; t < 2; t++) {
            int nt = kc * 2 + t;
            v8f ss = {}, so = {};
#pragma unroll
            for (int kt = 0; kt < 2; kt++) {
                size_t bidx = ((((size_t)bh * 2 + kt) * 32 + nt) * 32 + lane) * 16;
                AFrag k1f, k2f;
                k1f.q[0] = ((const uint4*)(k1p + bidx))[0];
                k1f.q[1] = ((const uint4*)(k1p + bidx))[1];
                k2f.q[0] = ((const uint4*)(k2p + bidx))[0];
                k2f.q[1] = ((const uint4*)(k2p + bidx))[1];
                ss = wmma_bf16(qa[kt], k1f, ss);
                so = wmma_bf16(qa[kt], k2f, so);
            }
            int lk   = nt * 16 + (lane & 15);
            int qmk  = (int)qmask[(size_t)b * LL + lk];
            bool kv  = (mask[(size_t)b * LL + lk] != 0);
#pragma unroll
            for (int i = 0; i < 8; i++) {
                float s = (qmq[i] == qmk) ? ss[i] : so[i];
                if (!kv) s = NEGV;
                if (g) {
                    float d = (float)((qt * 16 + hi8 + i) - lk);
                    s -= shift * d * d + bias;   // += -(shift*d^2 + bias_p)
                }
                S[t][i] = s;
            }
        }
        // V fragments for this key chunk (independent of softmax -> hoisted)
        AFrag vf[4];
#pragma unroll
        for (int d = 0; d < 4; d++) {
            size_t vidx = ((((size_t)bh * 16 + kc) * 4 + d) * 32 + lane) * 16;
            vf[d].q[0] = ((const uint4*)(vp + vidx))[0];
            vf[d].q[1] = ((const uint4*)(vp + vidx))[1];
        }
        // online softmax (row = 16 lanes within a half-wave for each element i)
#pragma unroll
        for (int i = 0; i < 8; i++) {
            float tm = fmaxf(S[0][i], S[1][i]);
            tm = fmaxf(tm, __shfl_xor(tm, 1, 32));
            tm = fmaxf(tm, __shfl_xor(tm, 2, 32));
            tm = fmaxf(tm, __shfl_xor(tm, 4, 32));
            tm = fmaxf(tm, __shfl_xor(tm, 8, 32));
            float nm = fmaxf(runmax[i], tm);
            float sc = __expf(runmax[i] - nm);
            runmax[i] = nm;
            float p0 = __expf(S[0][i] - nm);
            float p1 = __expf(S[1][i] - nm);
            runsum[i] = runsum[i] * sc + p0 + p1;
#pragma unroll
            for (int d = 0; d < 4; d++) O[d][i] *= sc;
            lds[wv][i + hi8][lane & 15]        = f2bf(p0);
            lds[wv][i + hi8][16 + (lane & 15)] = f2bf(p1);
        }
        // rebuild P as A-fragment (per-wave LDS region; same-wave DS ordering)
        AFrag pf;
        {
            const unsigned short* lrow = &lds[wv][lane & 15][0];
            pf.q[0] = *(const uint4*)(lrow + hi8);
            pf.q[1] = *(const uint4*)(lrow + 16 + hi8);
        }
#pragma unroll
        for (int d = 0; d < 4; d++)
            O[d] = wmma_bf16(pf, vf[d], O[d]);
    }

    // finalize: full row sums, normalize
#pragma unroll
    for (int i = 0; i < 8; i++) {
        float s = runsum[i];
        s += __shfl_xor(s, 1, 32);
        s += __shfl_xor(s, 2, 32);
        s += __shfl_xor(s, 4, 32);
        s += __shfl_xor(s, 8, 32);
        float inv = 1.0f / s;
#pragma unroll
        for (int d = 0; d < 4; d++) O[d][i] *= inv;
    }
    // store attention output [b][l][h*64+dh] as bf16 (A matrix for fc GEMM)
#pragma unroll
    for (int d2 = 0; d2 < 4; d2++) {
        int col = h * DHH + d2 * 16 + (lane & 15);
#pragma unroll
        for (int i = 0; i < 8; i++) {
            int l = qt * 16 + i + hi8;
            aout[((size_t)b * LL + l) * DD + col] = f2bf(O[d2][i]);
        }
    }
}

// ---------------------------------------------------------------------------
// Kernel 5: fc GEMM. out[8192,1024] = Abf[8192,1024] @ Wfc_pack + bfc (f32).
// 32x64 wave tile, double-buffered like gemm_qkv.
// ---------------------------------------------------------------------------
__global__ __launch_bounds__(128) void gemm_fc(const unsigned short* __restrict__ ab,
                                               const unsigned short* __restrict__ wp,
                                               const float* __restrict__ bfc,
                                               float* __restrict__ out) {
    int lane = threadIdx.x & 31;
    int wv   = threadIdx.x >> 5;
    int task = blockIdx.x * 4 + wv;   // 4096 = 256 mt * 16 nt64
    int mt   = task >> 4;
    int nt64 = task & 15;
    int hi   = (lane >= 16) ? 1 : 0;
    int hi8  = hi * 8;
    const unsigned short* abase0 = ab + (size_t)(mt * 32 + (lane & 15)) * DD;
    const unsigned short* abase1 = abase0 + (size_t)16 * DD;

    v8f acc[2][4] = {};
    AFrag a0[2], b0[4], a1[2], b1[4];
    load_a2(a0, abase0, abase1, 0, hi8);
    load_b4(b0, wp, 0, 64, nt64, lane);
    for (int k0 = 0; k0 < DD; k0 += 64) {
        load_a2(a1, abase0, abase1, k0 + 32, hi8);
        load_b4(b1, wp, (k0 >> 5) + 1, 64, nt64, lane);
        mma_8(acc, a0, b0);
        if (k0 + 64 < DD) {
            load_a2(a0, abase0, abase1, k0 + 64, hi8);
            load_b4(b0, wp, (k0 >> 5) + 2, 64, nt64, lane);
        }
        mma_8(acc, a1, b1);
    }
#pragma unroll
    for (int mh = 0; mh < 2; mh++) {
#pragma unroll
        for (int t = 0; t < 4; t++) {
            int ng = nt64 * 64 + t * 16 + (lane & 15);
            float bv = bfc[ng];
#pragma unroll
            for (int i = 0; i < 8; i++) {
                int mg = mt * 32 + mh * 16 + i + hi8;
                out[(size_t)mg * DD + ng] = acc[mh][t][i] + bv;
            }
        }
    }
}

// ---------------------------------------------------------------------------
// Host launcher
// ---------------------------------------------------------------------------
extern "C" void kernel_launch(void* const* d_in, const int* in_sizes, int n_in,
                              void* d_out, int out_size, void* d_ws, size_t ws_size,
                              hipStream_t stream) {
    (void)in_sizes; (void)n_in; (void)out_size; (void)ws_size;
    const float*     x      = (const float*)d_in[0];
    const long long* mask   = (const long long*)d_in[1];
    const long long* qmask  = (const long long*)d_in[2];
    const float*     Wqkv   = (const float*)d_in[3];
    const float*     Wfc    = (const float*)d_in[4];
    const float*     bfc    = (const float*)d_in[5];
    const float*     shift  = (const float*)d_in[6];
    const float*     bias_p = (const float*)d_in[7];
    const int*       useg   = (const int*)d_in[8];
    float*           out    = (float*)d_out;

    // workspace partition (u16 elements)
    unsigned short* w = (unsigned short*)d_ws;
    unsigned short* xb    = w;                               // 8192*1024
    unsigned short* wqkvp = xb    + (size_t)MTOT * DD;       // 1024*4096
    unsigned short* wfcp  = wqkvp + (size_t)DD * 4 * DD;     // 1024*1024
    unsigned short* qbf   = wfcp  + (size_t)DD * DD;         // B*H*L*DH
    unsigned short* k1p   = qbf   + (size_t)MTOT * DD;
    unsigned short* k2p   = k1p   + (size_t)MTOT * DD;
    unsigned short* vp    = k2p   + (size_t)MTOT * DD;
    unsigned short* aout  = vp    + (size_t)MTOT * DD;

    cvt_bf16<<<4096, 256, 0, stream>>>(x, xb, (MTOT * DD) / 4);
    pack_b<<<(32 * 256 * 32) / 256, 256, 0, stream>>>(Wqkv, wqkvp, 4 * DD);
    pack_b<<<(32 * 64 * 32) / 256, 256, 0, stream>>>(Wfc, wfcp, DD);
    gemm_qkv<<<4096, 128, 0, stream>>>(xb, wqkvp, qbf, k1p, k2p, vp);
    attn_kernel<<<2048, 128, 0, stream>>>(qbf, k1p, k2p, vp, mask, qmask,
                                          shift, bias_p, useg, aout);
    gemm_fc<<<1024, 128, 0, stream>>>(aout, wfcp, bfc, out);
}
